// Attention_Module_69303592288382
// MI455X (gfx1250) — matrix operand order repaired and checked by reading, not compiled, not run on hardware
//
#include <hip/hip_runtime.h>
#include <math.h>

#define B_SZ   1024
#define N_SZ   200
#define NP     208            // N padded to 13 * 16
#define D_SZ   128
#define H_SZ   4
#define C2     256            // 2*D
#define PAD_IDX 100000
#define WPAIRS 65536          // H*D*C2/2 u32 pairs per weight matrix

typedef __attribute__((ext_vector_type(16))) __bf16 v16bf;
typedef __attribute__((ext_vector_type(8)))  float  v8f;

union ABt { v16bf v; unsigned int u[8]; };

__device__ __forceinline__ unsigned int pack2bf(float a, float b) {
  unsigned int ua = __float_as_uint(a), ub = __float_as_uint(b);
  ua += 0x7FFFu + ((ua >> 16) & 1u);          // RNE round to bf16
  ub += 0x7FFFu + ((ub >> 16) & 1u);
  return (ua >> 16) | (ub & 0xFFFF0000u);
}
__device__ __forceinline__ unsigned short f2bf(float a) {
  unsigned int ua = __float_as_uint(a);
  ua += 0x7FFFu + ((ua >> 16) & 1u);
  return (unsigned short)(ua >> 16);
}
__device__ __forceinline__ float bf2f(unsigned short h) {
  return __uint_as_float(((unsigned int)h) << 16);
}

// ---- per-launch weight pre-conversion: Wq|Wk f32 -> packed bf16 in d_ws ----
__global__ __launch_bounds__(256, 1)
void conv_weights_kernel(const float* __restrict__ Wq,
                         const float* __restrict__ Wk,
                         unsigned int* __restrict__ ws) {
  int i = blockIdx.x * blockDim.x + threadIdx.x;   // 0 .. 2*WPAIRS-1
  const float* src = (i < WPAIRS) ? Wq : Wk;
  int p = (i < WPAIRS) ? i : (i - WPAIRS);
  float2 f = ((const float2*)src)[p];
  ws[i] = pack2bf(f.x, f.y);
}

template <bool PRE>
__global__ __launch_bounds__(256, 1)
void fused_graph_attn_kernel(const int* __restrict__ entity_pairs,
                             const int* __restrict__ left_conn,
                             const int* __restrict__ right_conn,
                             const float* __restrict__ emb,
                             const float* __restrict__ Wv,
                             const float* __restrict__ Wk,
                             const float* __restrict__ Wq,
                             const float* __restrict__ Ww,
                             const float* __restrict__ Whead,
                             const float* __restrict__ ln_w,
                             const float* __restrict__ ln_b,
                             const unsigned int* __restrict__ wbf,  // PRE: bf16 Wq|Wk
                             float* __restrict__ out) {
  __shared__ unsigned short concat_s[NP * C2];   // 106.5 KB bf16
  __shared__ unsigned short q_s[NP * D_SZ];      // 53.2 KB bf16
  __shared__ unsigned short k_s[NP * D_SZ];      // 53.2 KB bf16
  __shared__ float wsum[NP];
  __shared__ float maskv[NP];
  __shared__ float u_s[C2];
  __shared__ float yflat[H_SZ * D_SZ];
  __shared__ float ent_self_s[D_SZ];
  __shared__ float xvec[D_SZ];
  __shared__ float red_s[2];

  const int t    = threadIdx.x;
  const int lane = t & 31;
  const int wave = t >> 5;
  const int half = lane >> 4;
  const int l16  = lane & 15;
  const int side = blockIdx.x & 1;
  const int b    = blockIdx.x >> 1;
  const int* conn = side ? right_conn : left_conn;
  const float scale = 0.08838834764831845f;      // D^-0.5

  if (t < D_SZ) {
    int eidx = entity_pairs[b * 2 + side];
    ent_self_s[t] = emb[(size_t)eidx * D_SZ + t];
  }

  // ---- gather: concat[n] = [emb[rel_idx[n]], emb[ent_idx[n]]] as bf16 ----
  for (int n = wave; n < NP; n += 8) {
    unsigned int* dst = (unsigned int*)&concat_s[n * C2];
    if (n < N_SZ) {
      int rel = conn[((size_t)b * N_SZ + n) * 2 + 0];
      int ent = conn[((size_t)b * N_SZ + n) * 2 + 1];
      if (lane == 0) maskv[n] = (rel == PAD_IDX) ? 1.f : 0.f;
      float4 fr = ((const float4*)(emb + (size_t)rel * D_SZ))[lane];
      float4 fe = ((const float4*)(emb + (size_t)ent * D_SZ))[lane];
      dst[lane * 2 + 0]      = pack2bf(fr.x, fr.y);
      dst[lane * 2 + 1]      = pack2bf(fr.z, fr.w);
      dst[64 + lane * 2 + 0] = pack2bf(fe.x, fe.y);
      dst[64 + lane * 2 + 1] = pack2bf(fe.z, fe.w);
    } else {
      if (lane == 0) maskv[n] = 1.f;
      dst[lane * 2] = 0u; dst[lane * 2 + 1] = 0u;
      dst[64 + lane * 2] = 0u; dst[64 + lane * 2 + 1] = 0u;
    }
  }
  __syncthreads();

  for (int h = 0; h < H_SZ; ++h) {
    if (t < NP) wsum[t] = 0.f;
    __syncthreads();

    // ---- Q/K projections: (208x256 bf16) x (256x128) -> LDS bf16 ----
    for (int job = wave; job < 208; job += 8) {
      const int isK  = job >= 104;
      const int tile = isK ? job - 104 : job;
      const int n0 = (tile % 13) * 16;
      const int d0 = (tile / 13) * 16;
      const float*        Wsrc = (isK ? Wk : Wq) + (size_t)h * D_SZ * C2;
      const unsigned int* Wbfs = wbf + (isK ? WPAIRS : 0) + h * (D_SZ * C2 / 2);
      unsigned short* dst = isK ? k_s : q_s;
      v8f acc = {};
      #pragma unroll
      for (int ks = 0; ks < 8; ++ks) {
        ABt A, Bm;
        const int c0 = ks * 32;
        #pragma unroll
        for (int j = 0; j < 8; ++j) {
          // A 16x32 bf16 layout: contiguous k = 16*half + 2j
          int ca = c0 + (half << 4) + (j << 1);
          A.u[j] = *(const unsigned int*)&concat_s[(n0 + l16) * C2 + ca];
          if (PRE) {
            // packed bf16 weights: one dword per fragment register
            Bm.u[j] = Wbfs[(size_t)(d0 + l16) * (C2 / 2) + (c0 >> 1) + (half << 3) + j];
          } else {
            int cb = c0 + (half << 4) + (j << 1);
            const float* wp = Wsrc + (size_t)(d0 + l16) * C2 + cb;
            Bm.u[j] = pack2bf(wp[0], wp[1]);
          }
        }
        acc = __builtin_amdgcn_wmma_f32_16x16x32_bf16(
                  false, A.v, false, Bm.v, (short)0, acc, false, false);
      }
      #pragma unroll
      for (int j = 0; j < 8; ++j) {
        int row = j + (half << 3);                 // C layout: VGPRj -> M
        dst[(n0 + row) * D_SZ + d0 + l16] = f2bf(acc[j]);
      }
    }
    __syncthreads();

    // ---- scores S = K Q^T, mask, softmax over m, column-sum -> wsum ----
    for (int rb = wave; rb < 13; rb += 8) {
      const int n0 = rb * 16;
      const v8f vzero = {};
      v8f acc[13];
      #pragma unroll
      for (int tt = 0; tt < 13; ++tt) acc[tt] = vzero;

      #pragma unroll
      for (int ks = 0; ks < 4; ++ks) {             // k-step outer: A reused 13x
        const int c0 = ks * 32;
        ABt A;
        #pragma unroll
        for (int j = 0; j < 8; ++j) {
          int ca = c0 + (half << 4) + (j << 1);
          A.u[j] = *(const unsigned int*)&k_s[(n0 + l16) * D_SZ + ca];
        }
        #pragma unroll
        for (int tt = 0; tt < 13; ++tt) {
          ABt Bm;
          const int m0 = tt * 16;
          #pragma unroll
          for (int j = 0; j < 8; ++j) {
            int cb = c0 + (half << 4) + (j << 1);  // B[k=d][n=m] = Q[m][d]
            Bm.u[j] = *(const unsigned int*)&q_s[(m0 + l16) * D_SZ + cb];
          }
          acc[tt] = __builtin_amdgcn_wmma_f32_16x16x32_bf16(
                        false, A.v, false, Bm.v, (short)0, acc[tt], false, false);
        }
      }

      float rowm[8], rmax[8], rsum[8];
      #pragma unroll
      for (int j = 0; j < 8; ++j) {
        rowm[j] = maskv[n0 + j + (half << 3)];
        rmax[j] = -3.0e38f;
      }
      // pass 1: scale + mask, per-lane row max
      #pragma unroll
      for (int tt = 0; tt < 13; ++tt) {
        const int c = tt * 16 + l16;
        const float cm = maskv[c];
        const bool cpad = (c >= N_SZ);
        #pragma unroll
        for (int j = 0; j < 8; ++j) {
          float s = acc[tt][j] * scale;
          s = (rowm[j] > 0.5f || cm > 0.5f) ? -1.0e9f : s;
          s = cpad ? -1.0e30f : s;                 // padded cols vanish in exp
          acc[tt][j] = s;
          rmax[j] = fmaxf(rmax[j], s);
        }
      }
      #pragma unroll
      for (int j = 0; j < 8; ++j) {                // 16-lane row-max reduce
        float m = rmax[j];
        m = fmaxf(m, __shfl_xor(m, 1, 32));
        m = fmaxf(m, __shfl_xor(m, 2, 32));
        m = fmaxf(m, __shfl_xor(m, 4, 32));
        m = fmaxf(m, __shfl_xor(m, 8, 32));
        rmax[j] = m;
        rsum[j] = 0.f;
      }
      // pass 2: exp + row sums
      #pragma unroll
      for (int tt = 0; tt < 13; ++tt)
        #pragma unroll
        for (int j = 0; j < 8; ++j) {
          float e = __expf(acc[tt][j] - rmax[j]);
          acc[tt][j] = e;
          rsum[j] += e;
        }
      #pragma unroll
      for (int j = 0; j < 8; ++j) {
        float s2 = rsum[j];
        s2 += __shfl_xor(s2, 1, 32);
        s2 += __shfl_xor(s2, 2, 32);
        s2 += __shfl_xor(s2, 4, 32);
        s2 += __shfl_xor(s2, 8, 32);
        int nrow = n0 + j + (half << 3);
        rsum[j] = (nrow < N_SZ) ? (1.f / s2) : 0.f;   // drop padded rows
      }
      // pass 3: column sums of P into wsum
      #pragma unroll
      for (int tt = 0; tt < 13; ++tt) {
        float cs = 0.f;
        #pragma unroll
        for (int j = 0; j < 8; ++j) cs += acc[tt][j] * rsum[j];
        cs += __shfl_xor(cs, 16, 32);
        if (half == 0) atomicAdd(&wsum[tt * 16 + l16], cs);
      }
    }
    __syncthreads();

    // ---- u = wsum . concat  (256-vector) ----
    {
      float u = 0.f;
      for (int m = 0; m < N_SZ; ++m)
        u += wsum[m] * bf2f(concat_s[m * C2 + t]);
      u_s[t] = u;
    }
    __syncthreads();
    // ---- y_h = u @ Wv[h]^T  (V never materialized) ----
    if (t < D_SZ) {
      const float* wv = Wv + (size_t)h * D_SZ * C2 + (size_t)t * C2;
      float y = 0.f;
      #pragma unroll 4
      for (int c = 0; c < C2; ++c) y += u_s[c] * wv[c];
      yflat[h * D_SZ + t] = y;
    }
    __syncthreads();
  }

  // ---- epilogue: Ww contraction + head proj + relu + residual + LN ----
  if (t < 2) red_s[t] = 0.f;
  if (t < D_SZ) {
    const float* wwr = Ww + (size_t)t * (H_SZ * D_SZ);
    float a = 0.f;
    #pragma unroll 4
    for (int k2 = 0; k2 < H_SZ * D_SZ; ++k2) a += yflat[k2] * wwr[k2];
    const float* whr = Whead + (size_t)t * D_SZ;
    float hs = 0.f;
    #pragma unroll 4
    for (int c = 0; c < D_SZ; ++c) hs += ent_self_s[c] * whr[c];
    xvec[t] = fmaxf(a + hs, 0.f) + ent_self_s[t];
  }
  __syncthreads();
  if (t < D_SZ) {
    atomicAdd(&red_s[0], xvec[t]);
    atomicAdd(&red_s[1], xvec[t] * xvec[t]);
  }
  __syncthreads();
  if (t < D_SZ) {
    float mu  = red_s[0] * (1.f / 128.f);
    float var = red_s[1] * (1.f / 128.f) - mu * mu;
    float x = (xvec[t] - mu) * rsqrtf(var + 1e-5f);
    out[((size_t)b * 2 + side) * D_SZ + t] = x * ln_w[t] + ln_b[t];
  }
}

extern "C" void kernel_launch(void* const* d_in, const int* in_sizes, int n_in,
                              void* d_out, int out_size, void* d_ws, size_t ws_size,
                              hipStream_t stream) {
  (void)in_sizes; (void)n_in; (void)out_size;
  const int*   entity_pairs = (const int*)  d_in[0];
  const int*   left_conn    = (const int*)  d_in[1];
  const int*   right_conn   = (const int*)  d_in[3];
  const float* emb          = (const float*)d_in[5];
  const float* Wv           = (const float*)d_in[6];
  const float* Wk           = (const float*)d_in[7];
  const float* Wq           = (const float*)d_in[8];
  const float* Ww           = (const float*)d_in[9];
  const float* Whead        = (const float*)d_in[10];
  const float* ln_w         = (const float*)d_in[11];
  const float* ln_b         = (const float*)d_in[12];
  float* out = (float*)d_out;

  const size_t need = (size_t)2 * WPAIRS * sizeof(unsigned int); // 512 KB
  const bool pre = (d_ws != nullptr) && (ws_size >= need);

  dim3 grid(B_SZ * 2), block(256);
  if (pre) {
    unsigned int* ws = (unsigned int*)d_ws;
    conv_weights_kernel<<<dim3(2 * WPAIRS / 256), block, 0, stream>>>(Wq, Wk, ws);
    fused_graph_attn_kernel<true><<<grid, block, 0, stream>>>(
        entity_pairs, left_conn, right_conn, emb,
        Wv, Wk, Wq, Ww, Whead, ln_w, ln_b, ws, out);
  } else {
    fused_graph_attn_kernel<false><<<grid, block, 0, stream>>>(
        entity_pairs, left_conn, right_conn, emb,
        Wv, Wk, Wq, Ww, Whead, ln_w, ln_b, nullptr, out);
  }
}
